// SQUAT_46840913330667
// MI455X (gfx1250) — matrix operand order repaired and checked by reading, not compiled, not run on hardware
//
#include <hip/hip_runtime.h>
#include <hip/hip_bf16.h>

// ---------------------------------------------------------------------------
// Problem constants (from the reference)
// ---------------------------------------------------------------------------
#define BB    256      // batch
#define NN    12       // nodes
#define PP    66       // pairs
#define DNODE 1024     // node feature size
#define DEDGE 1024     // edge feature size
#define DD    512      // message size
#define D3    1536     // 3*D
#define H0    1024     // trunk hidden 0
#define H1    512      // trunk hidden 1
#define ROWS  (BB * PP)      // 16896
#define NODES (BB * NN)      // 3072
#define C_LR  9
#define C_CR  3
#define C_MR  5

typedef __attribute__((ext_vector_type(16))) __bf16 v16bf;
typedef __attribute__((ext_vector_type(8)))  __bf16 v8bf;
typedef __attribute__((ext_vector_type(8)))  float  v8f;

#ifndef __has_builtin
#define __has_builtin(x) 0
#endif

#if __has_builtin(__builtin_amdgcn_global_load_async_to_lds_b128)
#define HAS_ASYNC_LDS 1
#else
#define HAS_ASYNC_LDS 0
#endif

// 16-byte global -> LDS copy. Async (ASYNCcnt-tracked, bypasses VGPRs) when the
// gfx1250 builtin is available; plain staged copy otherwise.
__device__ __forceinline__ void async_copy16(const __bf16* g, __bf16* l) {
#if HAS_ASYNC_LDS
  // Builtin signature (from clang diagnostic): params are pointers to
  // int __attribute__((vector_size(16))) in AS1 (src) / AS3 (dst).
  typedef int v4i_vec __attribute__((vector_size(16)));
  typedef __attribute__((address_space(1))) v4i_vec gv4i;
  typedef __attribute__((address_space(3))) v4i_vec lv4i;
  // Generic LDS pointer truncates to the LDS byte offset (ISA 10.2 aperture rule).
  __builtin_amdgcn_global_load_async_to_lds_b128(
      (gv4i*)(size_t)g, (lv4i*)(unsigned)(size_t)l, 0, 0);
#else
  *(uint4*)l = *(const uint4*)g;
#endif
}

__device__ __forceinline__ void wait_async0() {
#if HAS_ASYNC_LDS
#if __has_builtin(__builtin_amdgcn_s_wait_asynccnt)
  __builtin_amdgcn_s_wait_asynccnt(0);
#else
  asm volatile("s_wait_asynccnt 0x0" ::: "memory");
#endif
#endif
}

// ---------------------------------------------------------------------------
// Weight transpose + fp32 -> bf16 convert: Wt[n][k] = bf16(W[k][n])
// ---------------------------------------------------------------------------
__global__ void k_transpose_cvt(const float* __restrict__ W,
                                __bf16* __restrict__ Wt, int K, int N) {
  int t = blockIdx.x * blockDim.x + threadIdx.x;
  if (t >= K * N) return;
  int k = t / N;
  int n = t - k * N;
  Wt[(size_t)n * K + k] = (__bf16)W[t];
}

// Concatenate three fp32 bias vectors of length `len` each.
__global__ void k_concat3(const float* __restrict__ a, const float* __restrict__ b,
                          const float* __restrict__ c, int len, float* __restrict__ out) {
  int t = blockIdx.x * blockDim.x + threadIdx.x;
  if (t >= 3 * len) return;
  int h = t / len, i = t - h * len;
  out[t] = (h == 0) ? a[i] : ((h == 1) ? b[i] : c[i]);
}

// ---------------------------------------------------------------------------
// Pair gather: fills ci[:,0:1024] (n1|n2), es = pair-summed edge features,
// embq = 0.25*(emb[i0,i1]+emb[i1,i0]), rowMask = valid ? 1 : 0
// One 256-thread block per (b,p) row.
// ---------------------------------------------------------------------------
__global__ __launch_bounds__(256)
void k_gather(const __bf16* __restrict__ nf,
              const float*  __restrict__ edge_features,
              const float*  __restrict__ edge_emb,
              const int*    __restrict__ pairs,
              const int*    __restrict__ num_rels,
              __bf16* __restrict__ ci,
              __bf16* __restrict__ es,
              float*  __restrict__ embq,
              float*  __restrict__ rowMask) {
  int row = blockIdx.x;
  int b = row / PP;
  int p = row - b * PP;
  int i0 = pairs[row * 2 + 0];
  int i1 = pairs[row * 2 + 1];
  bool valid = p < num_rels[b];
  int tid = threadIdx.x;

  const __bf16* n1 = nf + (size_t)(b * NN + i0) * DD;
  const __bf16* n2 = nf + (size_t)(b * NN + i1) * DD;
  __bf16* cir = ci + (size_t)row * D3;
  for (int j = tid; j < DD; j += 256) {
    cir[j]      = valid ? n1[j] : (__bf16)0.0f;
    cir[DD + j] = valid ? n2[j] : (__bf16)0.0f;
  }

  const float* ef01 = edge_features + (size_t)((b * NN + i0) * NN + i1) * DEDGE;
  const float* ef10 = edge_features + (size_t)((b * NN + i1) * NN + i0) * DEDGE;
  __bf16* esr = es + (size_t)row * DEDGE;
  for (int j = tid; j < DEDGE; j += 256) {
    esr[j] = (__bf16)(ef01[j] + ef10[j]);
  }

  const float* em01 = edge_emb + (size_t)((b * NN + i0) * NN + i1) * DD;
  const float* em10 = edge_emb + (size_t)((b * NN + i1) * NN + i0) * DD;
  float* eq = embq + (size_t)row * DD;
  for (int j = tid; j < DD; j += 256) {
    eq[j] = 0.25f * (em01[j] + em10[j]);
  }
  if (tid == 0) rowMask[row] = valid ? 1.0f : 0.0f;
}

// ---------------------------------------------------------------------------
// bf16 WMMA GEMM: C[M,N] = epilogue(alpha * A[M,K] @ B[K,N])
//   A: fp32 or bf16 row-major (lda);  B given pre-transposed bf16 Bt[N][K]
//   out = relu?( (alpha*acc + biasScale*bias[n] + extra[m][n]) * rowMask[m] )
// Block = 256 threads (8 wave32), tile 128x128, K-step 32, double-buffered
// LDS with async global->LDS copies (ASYNCcnt) when available.
// Wave grid 4(M) x 2(N); each wave: 2x4 tiles of v_wmma_f32_16x16x32_bf16.
// Fragment layouts follow CDNA5 ISA (16-bit A 16x32 / B 32x16).
// ---------------------------------------------------------------------------
#define TM 128
#define TN 128
#define TK 32
#define LDPAD 8

template<bool ABF16, bool RELU>
__global__ __launch_bounds__(256)
void k_gemm_wmma(const void* __restrict__ Avoid, int lda,
                 const __bf16* __restrict__ Bt, int K,
                 const float* __restrict__ bias, float biasScale,
                 float alpha,
                 const float* __restrict__ extra, int extraLd,
                 const float* __restrict__ rowMask,
                 __bf16* __restrict__ Cout, int ldo) {
  __shared__ __bf16 As[2][TM][TK + LDPAD];
  __shared__ __bf16 Bs[2][TN][TK + LDPAD];

  const int tid   = threadIdx.x;
  const int lane  = tid & 31;
  const int wid   = tid >> 5;
  const int waveM = wid & 3;      // 0..3 -> 32-row slab
  const int waveN = wid >> 2;     // 0..1 -> 64-col slab
  const int lhalf = lane >> 4;    // 0 | 1
  const int l15   = lane & 15;

  const int mBlk = blockIdx.x * TM;
  const int nBlk = blockIdx.y * TN;

  const int ldRow = tid >> 1;         // 0..127
  const int ldCol = (tid & 1) * 16;   // 0 | 16

  const float*  Af = (const float*)Avoid;
  const __bf16* Ab = (const __bf16*)Avoid;

  v8f acc[2][4];
  const v8f vzero = {0.f, 0.f, 0.f, 0.f, 0.f, 0.f, 0.f, 0.f};
#pragma unroll
  for (int r = 0; r < 2; ++r)
#pragma unroll
    for (int c = 0; c < 4; ++c) acc[r][c] = vzero;

  const int nk = K / TK;

  // ---- tile stager: tile kb -> LDS buffer `buf` ----
  auto stage = [&](int kb, int buf) {
    const int k0 = kb * TK;
    if (ABF16) {
      const __bf16* src = Ab + (size_t)(mBlk + ldRow) * lda + k0 + ldCol;
      async_copy16(src,     &As[buf][ldRow][ldCol]);
      async_copy16(src + 8, &As[buf][ldRow][ldCol + 8]);
    } else {
      const float* src = Af + (size_t)(mBlk + ldRow) * lda + k0 + ldCol;
#pragma unroll
      for (int i = 0; i < 16; i += 4) {
        float4 v = *(const float4*)(src + i);
        As[buf][ldRow][ldCol + i + 0] = (__bf16)v.x;
        As[buf][ldRow][ldCol + i + 1] = (__bf16)v.y;
        As[buf][ldRow][ldCol + i + 2] = (__bf16)v.z;
        As[buf][ldRow][ldCol + i + 3] = (__bf16)v.w;
      }
    }
    const __bf16* bsrc = Bt + (size_t)(nBlk + ldRow) * K + k0 + ldCol;
    async_copy16(bsrc,     &Bs[buf][ldRow][ldCol]);
    async_copy16(bsrc + 8, &Bs[buf][ldRow][ldCol + 8]);
    if (kb + 1 < nk) {
      __builtin_prefetch(bsrc + TK, 0, 1);   // global_prefetch_b8 next K-slab
    }
  };

  stage(0, 0);

  for (int kb = 0; kb < nk; ++kb) {
    const int cur = kb & 1;

    wait_async0();      // own async writes into buffer `cur` complete
    __syncthreads();    // all waves' writes visible; prior reads of `cur^1` done

    if (kb + 1 < nk) stage(kb + 1, cur ^ 1);  // overlap next copy with compute

    // ---- build fragments & issue WMMAs ----
    v16bf afrag[2], bfrag[4];
#pragma unroll
    for (int r = 0; r < 2; ++r) {
      const int aRow = waveM * 32 + r * 16 + l15;
      const int koff = lhalf * 8;           // lanes 0-15: K0..7/16..23 ; 16-31: K8..15/24..31
      v8bf lo = *(const v8bf*)&As[cur][aRow][koff];
      v8bf hi = *(const v8bf*)&As[cur][aRow][koff + 16];
      v16bf a;
#pragma unroll
      for (int i = 0; i < 8; ++i) { a[i] = lo[i]; a[8 + i] = hi[i]; }
      afrag[r] = a;
    }
#pragma unroll
    for (int c = 0; c < 4; ++c) {
      const int bRow = waveN * 64 + c * 16 + l15;   // Bt row = output column n
      const int koff = lhalf * 8;                   // lanes 0-15: K0..7/16..23 ; 16-31: K8..15/24..31
      v8bf lo = *(const v8bf*)&Bs[cur][bRow][koff];
      v8bf hi = *(const v8bf*)&Bs[cur][bRow][koff + 16];
      v16bf bfr;
#pragma unroll
      for (int i = 0; i < 8; ++i) { bfr[i] = lo[i]; bfr[8 + i] = hi[i]; }
      bfrag[c] = bfr;
    }
#pragma unroll
    for (int r = 0; r < 2; ++r)
#pragma unroll
      for (int c = 0; c < 4; ++c)
        acc[r][c] = __builtin_amdgcn_wmma_f32_16x16x32_bf16(
            false, afrag[r], false, bfrag[c], (short)0, acc[r][c], false, false);
  }

  // ---- epilogue: C/D layout -> lanes 0-15 rows M=0..7, 16-31 rows M=8..15 ----
#pragma unroll
  for (int r = 0; r < 2; ++r) {
#pragma unroll
    for (int c = 0; c < 4; ++c) {
      const int n = nBlk + waveN * 64 + c * 16 + l15;
      const float bv = bias ? bias[n] * biasScale : 0.0f;
#pragma unroll
      for (int e = 0; e < 8; ++e) {
        const int m = mBlk + waveM * 32 + r * 16 + lhalf * 8 + e;
        float v = alpha * acc[r][c][e] + bv;
        if (extra)   v += extra[(size_t)m * extraLd + n];
        if (rowMask) v *= rowMask[m];
        if (RELU)    v = v > 0.0f ? v : 0.0f;
        Cout[(size_t)m * ldo + n] = (__bf16)v;
      }
    }
  }
}

// ---------------------------------------------------------------------------
// Final tiny head layer: out[row,c] = sum_k h1[row,k] * W2[k,c] + b2[c]
// ---------------------------------------------------------------------------
__global__ void k_head(const __bf16* __restrict__ h1, const float* __restrict__ W2,
                       const float* __restrict__ b2, float* __restrict__ out,
                       int rows, int C) {
  int t = blockIdx.x * blockDim.x + threadIdx.x;
  if (t >= rows * C) return;
  int row = t / C;
  int c = t - row * C;
  const __bf16* hp = h1 + (size_t)row * H1;
  float s = 0.0f;
  for (int k = 0; k < H1; ++k) s += (float)hp[k] * W2[k * C + c];
  out[t] = s + b2[c];
}

// ---------------------------------------------------------------------------
extern "C" void kernel_launch(void* const* d_in, const int* in_sizes, int n_in,
                              void* d_out, int out_size, void* d_ws, size_t ws_size,
                              hipStream_t stream) {
  (void)in_sizes; (void)n_in; (void)out_size; (void)ws_size;

  const float* node_features   = (const float*)d_in[0];
  const float* edge_features   = (const float*)d_in[1];
  const float* edge_embeddings = (const float*)d_in[2];
  const int*   pairs           = (const int*)d_in[3];
  const int*   num_rels        = (const int*)d_in[4];
  const float* W_node = (const float*)d_in[5];
  const float* b_node = (const float*)d_in[6];
  const float* W_edge = (const float*)d_in[7];
  const float* b_edge = (const float*)d_in[8];
  const float* hW0[3] = { (const float*)d_in[9],  (const float*)d_in[15], (const float*)d_in[21] };
  const float* hb0[3] = { (const float*)d_in[10], (const float*)d_in[16], (const float*)d_in[22] };
  const float* hW1[3] = { (const float*)d_in[11], (const float*)d_in[17], (const float*)d_in[23] };
  const float* hb1[3] = { (const float*)d_in[12], (const float*)d_in[18], (const float*)d_in[24] };
  const float* hW2[3] = { (const float*)d_in[13], (const float*)d_in[19], (const float*)d_in[25] };
  const float* hb2[3] = { (const float*)d_in[14], (const float*)d_in[20], (const float*)d_in[26] };

  // ---- carve workspace ----
  char* ws = (char*)d_ws;
  size_t off = 0;
  auto carve = [&](size_t bytes) -> void* {
    void* p = ws + off;
    off = (off + bytes + 255) & ~(size_t)255;
    return p;
  };
  __bf16* Wnode_t = (__bf16*)carve((size_t)DD * DNODE * 2);
  __bf16* Wedge_t = (__bf16*)carve((size_t)DD * DEDGE * 2);
  __bf16* W0t     = (__bf16*)carve((size_t)3 * H0 * D3 * 2);     // [3072][1536]
  float*  b0cat   = (float*) carve((size_t)3 * H0 * 4);
  __bf16* W1t     = (__bf16*)carve((size_t)3 * H1 * H0 * 2);     // 3 x [512][1024]
  __bf16* nf      = (__bf16*)carve((size_t)NODES * DD * 2);
  __bf16* es      = (__bf16*)carve((size_t)ROWS * DEDGE * 2);
  float*  embq    = (float*) carve((size_t)ROWS * DD * 4);
  float*  rowMask = (float*) carve((size_t)ROWS * 4);
  __bf16* ci      = (__bf16*)carve((size_t)ROWS * D3 * 2);
  __bf16* h0      = (__bf16*)carve((size_t)ROWS * (3 * H0) * 2); // [16896][3072]
  __bf16* h1      = (__bf16*)carve((size_t)3 * ROWS * H1 * 2);

  // ---- weight conversion / transposition ----
  k_transpose_cvt<<<(DNODE * DD + 255) / 256, 256, 0, stream>>>(W_node, Wnode_t, DNODE, DD);
  k_transpose_cvt<<<(DEDGE * DD + 255) / 256, 256, 0, stream>>>(W_edge, Wedge_t, DEDGE, DD);
  for (int h = 0; h < 3; ++h) {
    k_transpose_cvt<<<(D3 * H0 + 255) / 256, 256, 0, stream>>>(
        hW0[h], W0t + (size_t)h * H0 * D3, D3, H0);
    k_transpose_cvt<<<(H0 * H1 + 255) / 256, 256, 0, stream>>>(
        hW1[h], W1t + (size_t)h * H1 * H0, H0, H1);
  }
  k_concat3<<<(3 * H0 + 255) / 256, 256, 0, stream>>>(hb0[0], hb0[1], hb0[2], H0, b0cat);

  // ---- nf = node_features @ W_node + b_node  (fp32 A -> bf16 out) ----
  {
    dim3 g(NODES / TM, DD / TN);
    k_gemm_wmma<false, false><<<g, 256, 0, stream>>>(
        node_features, DNODE, Wnode_t, DNODE, b_node, 1.0f, 1.0f,
        nullptr, 0, nullptr, nf, DD);
  }

  // ---- gather pairs -> ci[:,0:1024], es, embq, rowMask ----
  k_gather<<<ROWS, 256, 0, stream>>>(nf, edge_features, edge_embeddings,
                                     pairs, num_rels, ci, es, embq, rowMask);

  // ---- ci[:,1024:1536] = mask * (0.25*es@W_edge + embq + 0.5*b_edge) ----
  {
    dim3 g(ROWS / TM, DD / TN);
    k_gemm_wmma<true, false><<<g, 256, 0, stream>>>(
        es, DEDGE, Wedge_t, DEDGE, b_edge, 0.5f, 0.25f,
        embq, DD, rowMask, ci + 1024, D3);
  }

  // ---- trunk layer 0 (all 3 heads concatenated, N = 3072, ReLU) ----
  {
    dim3 g(ROWS / TM, (3 * H0) / TN);
    k_gemm_wmma<true, true><<<g, 256, 0, stream>>>(
        ci, D3, W0t, D3, b0cat, 1.0f, 1.0f,
        nullptr, 0, nullptr, h0, 3 * H0);
  }

  // ---- trunk layer 1 per head (ReLU) ----
  {
    dim3 g(ROWS / TM, H1 / TN);
    for (int h = 0; h < 3; ++h) {
      k_gemm_wmma<true, true><<<g, 256, 0, stream>>>(
          h0 + (size_t)h * H0, 3 * H0, W1t + (size_t)h * H1 * H0, H0,
          hb1[h], 1.0f, 1.0f, nullptr, 0, nullptr,
          h1 + (size_t)h * ROWS * H1, H1);
    }
  }

  // ---- final tiny layers -> d_out (lr | cr | mr, fp32) ----
  float* out_lr = (float*)d_out;
  float* out_cr = out_lr + (size_t)ROWS * C_LR;
  float* out_mr = out_cr + (size_t)ROWS * C_CR;
  k_head<<<(ROWS * C_LR + 255) / 256, 256, 0, stream>>>(h1,                          hW2[0], hb2[0], out_lr, ROWS, C_LR);
  k_head<<<(ROWS * C_CR + 255) / 256, 256, 0, stream>>>(h1 + (size_t)ROWS * H1,      hW2[1], hb2[1], out_cr, ROWS, C_CR);
  k_head<<<(ROWS * C_MR + 255) / 256, 256, 0, stream>>>(h1 + (size_t)2 * ROWS * H1,  hW2[2], hb2[2], out_mr, ROWS, C_MR);
}
